// BasicTransformerBlock_42142219108974
// MI455X (gfx1250) — matrix operand order repaired and checked by reading, not compiled, not run on hardware
//
#include <hip/hip_runtime.h>
#include <math.h>

typedef __bf16 bf16_t;
typedef __bf16 v16bf __attribute__((ext_vector_type(16)));
typedef __bf16 v8bf  __attribute__((ext_vector_type(8)));
typedef float  v8f   __attribute__((ext_vector_type(8)));

static __device__ inline v8f zero8() {
  v8f z;
#pragma unroll
  for (int i = 0; i < 8; ++i) z[i] = 0.f;
  return z;
}

static __device__ inline v8f wmma_bf16(v16bf a, v16bf b, v8f c) {
  return __builtin_amdgcn_wmma_f32_16x16x32_bf16(false, a, false, b, (short)0, c,
                                                 false, false);
}

union FragU { v16bf v; v8bf h2[2]; };

// ---------------------------------------------------------------------------
// LayerNorm: one block (256 threads) per row of 512.
// ---------------------------------------------------------------------------
__global__ __launch_bounds__(256) void ln_kernel(const float* __restrict__ x,
                                                 const float* __restrict__ g,
                                                 const float* __restrict__ b,
                                                 float* __restrict__ out) {
  __shared__ float red[256];
  const int row = blockIdx.x;
  const int tid = threadIdx.x;
  const size_t base = (size_t)row * 512;
  const float x0 = x[base + tid];
  const float x1 = x[base + 256 + tid];

  red[tid] = x0 + x1;
  __syncthreads();
#pragma unroll
  for (int s = 128; s > 0; s >>= 1) {
    if (tid < s) red[tid] += red[tid + s];
    __syncthreads();
  }
  const float mean = red[0] * (1.f / 512.f);
  __syncthreads();

  const float d0 = x0 - mean, d1 = x1 - mean;
  red[tid] = d0 * d0 + d1 * d1;
  __syncthreads();
#pragma unroll
  for (int s = 128; s > 0; s >>= 1) {
    if (tid < s) red[tid] += red[tid + s];
    __syncthreads();
  }
  const float rstd = rsqrtf(red[0] * (1.f / 512.f) + 1e-5f);

  out[base + tid]       = d0 * rstd * g[tid] + b[tid];
  out[base + 256 + tid] = d1 * rstd * g[256 + tid] + b[256 + tid];
}

// ---------------------------------------------------------------------------
// Tiled WMMA GEMM: C[M,N] = A[M,K] @ B[K,N] (+ bias + residual, + jagged mask)
// Block tile 128x128, K-step 32, 8 waves each own a 32x64 patch.
// EPI: 0 = plain, 1 = +bias+residual, 2 = +bias+residual, zero padded rows.
// ---------------------------------------------------------------------------
template <int EPI>
__global__ __launch_bounds__(256) void gemm_kernel(
    const float* __restrict__ A, const float* __restrict__ Bw,
    const float* __restrict__ bias, const float* __restrict__ res,
    float* __restrict__ C, int M, int Nn, int K,
    const int* __restrict__ lengths, int n_per_b) {
  __shared__ __align__(32) bf16_t ldsA[128 * 32];
  __shared__ __align__(32) bf16_t ldsB[32 * 128];

  const int tid = threadIdx.x;
  const int wave = tid >> 5, lane = tid & 31;
  const int lh = lane >> 4, ll = lane & 15;
  const int m0 = blockIdx.y * 128;
  const int n0 = blockIdx.x * 128;
  const int wm = (wave >> 1) * 32;  // 0,32,64,96
  const int wn = (wave & 1) * 64;   // 0,64

  v8f acc[2][4];
#pragma unroll
  for (int i = 0; i < 2; ++i)
#pragma unroll
    for (int j = 0; j < 4; ++j) acc[i][j] = zero8();

  for (int k0 = 0; k0 < K; k0 += 32) {
    __syncthreads();
    {  // stage A 128x32 (fp32 -> bf16)
      const int row = tid >> 1, c0 = (tid & 1) * 16;
      const float* src = A + (size_t)(m0 + row) * K + k0 + c0;
#pragma unroll
      for (int i = 0; i < 16; ++i) ldsA[row * 32 + c0 + i] = (bf16_t)src[i];
    }
    {  // stage B 32x128
      const int row = tid >> 3, c0 = (tid & 7) * 16;
      const float* src = Bw + (size_t)(k0 + row) * Nn + n0 + c0;
#pragma unroll
      for (int i = 0; i < 16; ++i) ldsB[row * 128 + c0 + i] = (bf16_t)src[i];
    }
    __syncthreads();

    v16bf af[2];
#pragma unroll
    for (int i2 = 0; i2 < 2; ++i2) {
      FragU u;
      const int row = wm + i2 * 16 + ll;
      const int kb = lh * 8;
      u.h2[0] = *(const v8bf*)&ldsA[row * 32 + kb];
      u.h2[1] = *(const v8bf*)&ldsA[row * 32 + kb + 16];
      af[i2] = u.v;
    }
#pragma unroll
    for (int j = 0; j < 4; ++j) {
      v16bf bfr;
#pragma unroll
      for (int s = 0; s < 16; ++s)
        bfr[s] = ldsB[(lh * 8 + (s & 7) + (s >> 3) * 16) * 128 + wn + j * 16 + ll];
#pragma unroll
      for (int i2 = 0; i2 < 2; ++i2) acc[i2][j] = wmma_bf16(af[i2], bfr, acc[i2][j]);
    }
  }

#pragma unroll
  for (int i2 = 0; i2 < 2; ++i2)
#pragma unroll
    for (int j = 0; j < 4; ++j)
#pragma unroll
      for (int r = 0; r < 8; ++r) {
        const int row = m0 + wm + i2 * 16 + r + 8 * lh;
        const int col = n0 + wn + j * 16 + ll;
        float v = acc[i2][j][r];
        if constexpr (EPI >= 1) v += bias[col] + res[(size_t)row * Nn + col];
        if constexpr (EPI == 2) {
          const int bb = row / n_per_b, nn = row % n_per_b;
          if (nn >= lengths[bb]) v = 0.f;
        }
        C[(size_t)row * Nn + col] = v;
      }
}

// ---------------------------------------------------------------------------
// Fused GEGLU: out[M,FF] = (A@W[:, :FF] + b_a) * gelu(A@W[:, FF:] + b_g)
// W is [K, 2*FF]; same A tile feeds both B panels.
// ---------------------------------------------------------------------------
__global__ __launch_bounds__(256) void geglu_kernel(
    const float* __restrict__ A, const float* __restrict__ Bw,
    const float* __restrict__ bias, float* __restrict__ Out, int M, int K,
    int FFn) {
  __shared__ __align__(32) bf16_t ldsA[128 * 32];
  __shared__ __align__(32) bf16_t ldsBa[32 * 128];
  __shared__ __align__(32) bf16_t ldsBg[32 * 128];

  const int tid = threadIdx.x;
  const int wave = tid >> 5, lane = tid & 31;
  const int lh = lane >> 4, ll = lane & 15;
  const int m0 = blockIdx.y * 128;
  const int n0 = blockIdx.x * 128;
  const int wm = (wave >> 1) * 32;
  const int wn = (wave & 1) * 64;
  const int ldb = 2 * FFn;

  v8f acc_a[2][4], acc_g[2][4];
#pragma unroll
  for (int i = 0; i < 2; ++i)
#pragma unroll
    for (int j = 0; j < 4; ++j) {
      acc_a[i][j] = zero8();
      acc_g[i][j] = zero8();
    }

  for (int k0 = 0; k0 < K; k0 += 32) {
    __syncthreads();
    {
      const int row = tid >> 1, c0 = (tid & 1) * 16;
      const float* src = A + (size_t)(m0 + row) * K + k0 + c0;
#pragma unroll
      for (int i = 0; i < 16; ++i) ldsA[row * 32 + c0 + i] = (bf16_t)src[i];
    }
    {
      const int row = tid >> 3, c0 = (tid & 7) * 16;
      const float* sa = Bw + (size_t)(k0 + row) * ldb + n0 + c0;
      const float* sg = sa + FFn;
#pragma unroll
      for (int i = 0; i < 16; ++i) {
        ldsBa[row * 128 + c0 + i] = (bf16_t)sa[i];
        ldsBg[row * 128 + c0 + i] = (bf16_t)sg[i];
      }
    }
    __syncthreads();

    v16bf af[2];
#pragma unroll
    for (int i2 = 0; i2 < 2; ++i2) {
      FragU u;
      const int row = wm + i2 * 16 + ll;
      const int kb = lh * 8;
      u.h2[0] = *(const v8bf*)&ldsA[row * 32 + kb];
      u.h2[1] = *(const v8bf*)&ldsA[row * 32 + kb + 16];
      af[i2] = u.v;
    }
#pragma unroll
    for (int j = 0; j < 4; ++j) {
      v16bf ba, bg;
#pragma unroll
      for (int s = 0; s < 16; ++s) {
        const int kk = lh * 8 + (s & 7) + (s >> 3) * 16;
        ba[s] = ldsBa[kk * 128 + wn + j * 16 + ll];
        bg[s] = ldsBg[kk * 128 + wn + j * 16 + ll];
      }
#pragma unroll
      for (int i2 = 0; i2 < 2; ++i2) {
        acc_a[i2][j] = wmma_bf16(af[i2], ba, acc_a[i2][j]);
        acc_g[i2][j] = wmma_bf16(af[i2], bg, acc_g[i2][j]);
      }
    }
  }

#pragma unroll
  for (int i2 = 0; i2 < 2; ++i2)
#pragma unroll
    for (int j = 0; j < 4; ++j)
#pragma unroll
      for (int r = 0; r < 8; ++r) {
        const int row = m0 + wm + i2 * 16 + r + 8 * lh;
        const int col = n0 + wn + j * 16 + ll;
        const float pa = acc_a[i2][j][r] + bias[col];
        const float pg = acc_g[i2][j][r] + bias[FFn + col];
        const float ge = 0.5f * pg * (1.0f + erff(pg * 0.70710678118f));
        Out[(size_t)row * FFn + col] = pa * ge;
      }
}

// ---------------------------------------------------------------------------
// Flash attention (bf16 WMMA, online softmax). One block = (b,h) x 128 Q rows.
// Q/K/V are [B, Nq|Nk, 512] fp32; head h occupies cols h*64..h*64+63.
// ---------------------------------------------------------------------------
__global__ __launch_bounds__(256) void flash_attn_kernel(
    const float* __restrict__ Qg, const float* __restrict__ Kg,
    const float* __restrict__ Vg, float* __restrict__ Og,
    const int* __restrict__ lengths, int Nq, int Nk) {
  __shared__ __align__(32) bf16_t ldsQ[128 * 64];
  __shared__ __align__(32) bf16_t ldsKt[64 * 32];  // [d][key]
  __shared__ __align__(32) bf16_t ldsV[32 * 64];   // [key][d]
  __shared__ __align__(32) bf16_t ldsP[8][16 * 32];

  const int tid = threadIdx.x;
  const int wave = tid >> 5, lane = tid & 31;
  const int lh = lane >> 4, ll = lane & 15;
  const int b = blockIdx.y >> 3;
  const int h = blockIdx.y & 7;
  const int q0 = blockIdx.x * 128;
  const int len = lengths ? lengths[b] : Nk;

  {  // stage Q 128x64 -> bf16
    const int row = tid >> 1, d0 = (tid & 1) * 32;
    const float* src = Qg + ((size_t)b * Nq + q0 + row) * 512 + h * 64 + d0;
#pragma unroll
    for (int i = 0; i < 32; ++i) ldsQ[row * 64 + d0 + i] = (bf16_t)src[i];
  }
  __syncthreads();

  v16bf qf[2];
  {
    const int row = wave * 16 + ll;
    const int kb = lh * 8;
#pragma unroll
    for (int c = 0; c < 2; ++c) {
      FragU u;
      u.h2[0] = *(const v8bf*)&ldsQ[row * 64 + c * 32 + kb];
      u.h2[1] = *(const v8bf*)&ldsQ[row * 64 + c * 32 + kb + 16];
      qf[c] = u.v;
    }
  }

  v8f oacc[4];
#pragma unroll
  for (int j = 0; j < 4; ++j) oacc[j] = zero8();
  float mrun[8], lrun[8];
#pragma unroll
  for (int r = 0; r < 8; ++r) {
    mrun[r] = -3.0e38f;
    lrun[r] = 0.f;
  }

  const int nkb = Nk / 32;
  for (int kb = 0; kb < nkb; ++kb) {
    const int key0 = kb * 32;
    __syncthreads();
    {  // stage K transposed: ldsKt[d][key]
      const int key = tid & 31;
      const int d0 = (tid >> 5) * 8;
      const float* src = Kg + ((size_t)b * Nk + key0 + key) * 512 + h * 64 + d0;
#pragma unroll
      for (int i = 0; i < 8; ++i) ldsKt[(d0 + i) * 32 + key] = (bf16_t)src[i];
    }
    {  // stage V: ldsV[key][d]
      const int key = tid >> 3;
      const int d0 = (tid & 7) * 8;
      const float* src = Vg + ((size_t)b * Nk + key0 + key) * 512 + h * 64 + d0;
#pragma unroll
      for (int i = 0; i < 8; ++i) ldsV[key * 64 + d0 + i] = (bf16_t)src[i];
    }
    __syncthreads();

    // S = Q @ K^T : two 16x16 tiles of (16 rows x 32 keys)
    v8f s[2];
#pragma unroll
    for (int t = 0; t < 2; ++t) {
      s[t] = zero8();
#pragma unroll
      for (int c = 0; c < 2; ++c) {
        v16bf kf;
#pragma unroll
        for (int s2 = 0; s2 < 16; ++s2)
          kf[s2] = ldsKt[(c * 32 + lh * 8 + (s2 & 7) + (s2 >> 3) * 16) * 32 + t * 16 + ll];
        s[t] = wmma_bf16(qf[c], kf, s[t]);
      }
    }

    // scale, mask, online softmax (lane = key column within 16-tile)
    bool km[2];
#pragma unroll
    for (int t = 0; t < 2; ++t) km[t] = (key0 + t * 16 + ll) >= len;

#pragma unroll
    for (int r = 0; r < 8; ++r) {
      float mx = -3.0e38f;
#pragma unroll
      for (int t = 0; t < 2; ++t) {
        float v = s[t][r] * 0.125f;
        v = km[t] ? -3.0e38f : v;
        s[t][r] = v;
        mx = fmaxf(mx, v);
      }
#pragma unroll
      for (int off = 1; off < 16; off <<= 1) mx = fmaxf(mx, __shfl_xor(mx, off, 32));
      const float newm = fmaxf(mrun[r], mx);
      const float alpha = __expf(mrun[r] - newm);
      float rs = 0.f;
#pragma unroll
      for (int t = 0; t < 2; ++t) {
        const float p = km[t] ? 0.f : __expf(s[t][r] - newm);
        s[t][r] = p;
        rs += p;
      }
#pragma unroll
      for (int off = 1; off < 16; off <<= 1) rs += __shfl_xor(rs, off, 32);
      lrun[r] = lrun[r] * alpha + rs;
      mrun[r] = newm;
#pragma unroll
      for (int j = 0; j < 4; ++j) oacc[j][r] *= alpha;
    }

    // C-layout P -> wave-private LDS -> A-layout fragment
#pragma unroll
    for (int t = 0; t < 2; ++t)
#pragma unroll
      for (int r = 0; r < 8; ++r)
        ldsP[wave][(r + 8 * lh) * 32 + t * 16 + ll] = (bf16_t)s[t][r];
    asm volatile("s_wait_dscnt 0" ::: "memory");

    FragU up;
    {
      const int row = ll;
      const int kb2 = lh * 8;
      up.h2[0] = *(const v8bf*)&ldsP[wave][row * 32 + kb2];
      up.h2[1] = *(const v8bf*)&ldsP[wave][row * 32 + kb2 + 16];
    }
#pragma unroll
    for (int j = 0; j < 4; ++j) {
      v16bf vf;
#pragma unroll
      for (int s2 = 0; s2 < 16; ++s2)
        vf[s2] = ldsV[(lh * 8 + (s2 & 7) + (s2 >> 3) * 16) * 64 + j * 16 + ll];
      oacc[j] = wmma_bf16(up.v, vf, oacc[j]);
    }
  }

#pragma unroll
  for (int r = 0; r < 8; ++r) {
    const float inv = lrun[r] > 0.f ? 1.f / lrun[r] : 0.f;
#pragma unroll
    for (int j = 0; j < 4; ++j) oacc[j][r] *= inv;
  }
#pragma unroll
  for (int j = 0; j < 4; ++j)
#pragma unroll
    for (int r = 0; r < 8; ++r) {
      const int row = q0 + wave * 16 + r + 8 * lh;
      const int d = j * 16 + ll;
      Og[((size_t)b * Nq + row) * 512 + h * 64 + d] = oacc[j][r];
    }
}

// ---------------------------------------------------------------------------
extern "C" void kernel_launch(void* const* d_in, const int* in_sizes, int n_in,
                              void* d_out, int out_size, void* d_ws,
                              size_t ws_size, hipStream_t stream) {
  (void)in_sizes; (void)n_in; (void)out_size; (void)ws_size;
  const float* x    = (const float*)d_in[0];
  const float* ctx  = (const float*)d_in[1];
  const int*   len  = (const int*)d_in[2];
  const float* wq1  = (const float*)d_in[3];
  const float* wk1  = (const float*)d_in[4];
  const float* wv1  = (const float*)d_in[5];
  const float* wo1  = (const float*)d_in[6];
  const float* bo1  = (const float*)d_in[7];
  const float* wq2  = (const float*)d_in[8];
  const float* wk2  = (const float*)d_in[9];
  const float* wv2  = (const float*)d_in[10];
  const float* wo2  = (const float*)d_in[11];
  const float* bo2  = (const float*)d_in[12];
  const float* wffi = (const float*)d_in[13];
  const float* bffi = (const float*)d_in[14];
  const float* wffo = (const float*)d_in[15];
  const float* bffo = (const float*)d_in[16];
  const float* g1 = (const float*)d_in[17];
  const float* b1 = (const float*)d_in[18];
  const float* g2 = (const float*)d_in[19];
  const float* b2 = (const float*)d_in[20];
  const float* g3 = (const float*)d_in[21];
  const float* b3 = (const float*)d_in[22];

  float* xo = (float*)d_out;
  float* ws = (float*)d_ws;
  float* h  = ws;                 // 8.4M floats
  float* q  = ws + 8388608;       // 8.4M
  float* k  = ws + 16777216;      // 8.4M
  float* v  = ws + 25165824;      // 8.4M
  float* ao = ws + 33554432;      // 8.4M
  float* hf = q;                  // reuse q|k|v|ao region: 33.5M floats

  const dim3 blk(256);
  const int BN = 8 * 2048, BS = 8 * 256;
  const dim3 gproj(512 / 128, BN / 128);
  const dim3 gkv(512 / 128, BS / 128);
  const dim3 gfa(2048 / 128, 64);
  const dim3 gff(2048 / 128, BN / 128);

  // --- attn1 (self, jagged-masked) ---
  ln_kernel<<<BN, blk, 0, stream>>>(x, g1, b1, h);
  gemm_kernel<0><<<gproj, blk, 0, stream>>>(h, wq1, nullptr, nullptr, q, BN, 512, 512, nullptr, 0);
  gemm_kernel<0><<<gproj, blk, 0, stream>>>(h, wk1, nullptr, nullptr, k, BN, 512, 512, nullptr, 0);
  gemm_kernel<0><<<gproj, blk, 0, stream>>>(h, wv1, nullptr, nullptr, v, BN, 512, 512, nullptr, 0);
  flash_attn_kernel<<<gfa, blk, 0, stream>>>(q, k, v, ao, len, 2048, 2048);
  gemm_kernel<1><<<gproj, blk, 0, stream>>>(ao, wo1, bo1, x, xo, BN, 512, 512, nullptr, 0);

  // --- attn2 (cross to dense context) ---
  ln_kernel<<<BN, blk, 0, stream>>>(xo, g2, b2, h);
  gemm_kernel<0><<<gproj, blk, 0, stream>>>(h, wq2, nullptr, nullptr, q, BN, 512, 512, nullptr, 0);
  gemm_kernel<0><<<gkv, blk, 0, stream>>>(ctx, wk2, nullptr, nullptr, k, BS, 512, 768, nullptr, 0);
  gemm_kernel<0><<<gkv, blk, 0, stream>>>(ctx, wv2, nullptr, nullptr, v, BS, 512, 768, nullptr, 0);
  flash_attn_kernel<<<gfa, blk, 0, stream>>>(q, k, v, ao, nullptr, 2048, 256);
  gemm_kernel<1><<<gproj, blk, 0, stream>>>(ao, wo2, bo2, xo, xo, BN, 512, 512, nullptr, 0);

  // --- GEGLU FF + final residual + jagged zeroing ---
  ln_kernel<<<BN, blk, 0, stream>>>(xo, g3, b3, h);
  geglu_kernel<<<gff, blk, 0, stream>>>(h, wffi, bffi, hf, BN, 512, 2048);
  gemm_kernel<2><<<gproj, blk, 0, stream>>>(hf, wffo, bffo, xo, xo, BN, 512, 2048, len, 2048);
}
